// Attention_Separate_79903571574945
// MI455X (gfx1250) — matrix operand order repaired and checked by reading, not compiled, not run on hardware
//
#include <hip/hip_runtime.h>
#include <hip/hip_bf16.h>

typedef __bf16 v16bf __attribute__((ext_vector_type(16)));
typedef float  v8f   __attribute__((ext_vector_type(8)));
typedef unsigned short su8 __attribute__((ext_vector_type(8)));
typedef unsigned short u16;

union BfVec {
    v16bf v;
    su8   h[2];
    u16   u[16];
};

#define NB   2
#define NN   2048
#define NH   8
#define NR   64
#define ND   1024
#define BN   (NB * NN)        // 4096
#define QKC  (NH * NR)        // 512
#define VC   (NH * ND)        // 8192

__device__ __forceinline__ u16 f2bf(float f) {
    unsigned int u = __float_as_uint(f);
    unsigned int r = u + 0x7FFFu + ((u >> 16) & 1u);
    return (u16)(r >> 16);
}

__device__ __forceinline__ v8f wmma_bf16(const BfVec& a, const BfVec& b, v8f c) {
    return __builtin_amdgcn_wmma_f32_16x16x32_bf16(false, a.v, false, b.v,
                                                   (short)0, c, false, false);
}

// ---------------- pass 0: f32 -> bf16 conversion ----------------
__global__ void k_convert(const float* __restrict__ s, u16* __restrict__ d, int n) {
    for (int i = blockIdx.x * blockDim.x + threadIdx.x; i < n;
         i += gridDim.x * blockDim.x)
        d[i] = f2bf(s[i]);
}

// ---- pass 1: C[m,n] = sum_k A[m,k] * W[n,k], 2x4 register-blocked ---------
// A: [M x K] bf16 row-major, W: [Ncols x K] bf16 row-major, C: [M x Ncols] bf16
// wave computes a 32x64 macro-tile (8 WMMA accumulators)
__global__ __launch_bounds__(256) void k_gemm(const u16* __restrict__ A,
                                              const u16* __restrict__ W,
                                              u16* __restrict__ C,
                                              int Ncols, int K) {
    const int wave = threadIdx.x >> 5;
    const int lane = threadIdx.x & 31;
    const int lm   = lane & 15;
    const int half = lane >> 4;
    const int lhA  = half * 8;      // A operand K sub-base
    const int lhB  = half * 16;     // unused; B loads also use lhA

    const int tilesN = Ncols >> 6;                  // 64-wide macro columns
    const int id     = blockIdx.x * 8 + wave;
    const int tm     = (id / tilesN) << 5;          // 32 rows
    const int tn     = (id % tilesN) << 6;          // 64 cols

    const u16* a0p = A + (tm + lm) * K;
    const u16* a1p = A + (tm + 16 + lm) * K;
    const u16* w0p = W + (tn + lm) * K;
    const u16* w1p = W + (tn + 16 + lm) * K;
    const u16* w2p = W + (tn + 32 + lm) * K;
    const u16* w3p = W + (tn + 48 + lm) * K;

    v8f acc[2][4];
#pragma unroll
    for (int i = 0; i < 2; i++)
#pragma unroll
        for (int j = 0; j < 4; j++) acc[i][j] = {};

    for (int k0 = 0; k0 < K; k0 += 32) {
        BfVec a0, a1, b0, b1, b2, b3;
        a0.h[0] = *(const su8*)(a0p + k0 + lhA);
        a0.h[1] = *(const su8*)(a0p + k0 + 16 + lhA);
        a1.h[0] = *(const su8*)(a1p + k0 + lhA);
        a1.h[1] = *(const su8*)(a1p + k0 + 16 + lhA);
        b0.h[0] = *(const su8*)(w0p + k0 + lhA);
        b0.h[1] = *(const su8*)(w0p + k0 + 16 + lhA);
        b1.h[0] = *(const su8*)(w1p + k0 + lhA);
        b1.h[1] = *(const su8*)(w1p + k0 + 16 + lhA);
        b2.h[0] = *(const su8*)(w2p + k0 + lhA);
        b2.h[1] = *(const su8*)(w2p + k0 + 16 + lhA);
        b3.h[0] = *(const su8*)(w3p + k0 + lhA);
        b3.h[1] = *(const su8*)(w3p + k0 + 16 + lhA);
        acc[0][0] = wmma_bf16(a0, b0, acc[0][0]);
        acc[0][1] = wmma_bf16(a0, b1, acc[0][1]);
        acc[0][2] = wmma_bf16(a0, b2, acc[0][2]);
        acc[0][3] = wmma_bf16(a0, b3, acc[0][3]);
        acc[1][0] = wmma_bf16(a1, b0, acc[1][0]);
        acc[1][1] = wmma_bf16(a1, b1, acc[1][1]);
        acc[1][2] = wmma_bf16(a1, b2, acc[1][2]);
        acc[1][3] = wmma_bf16(a1, b3, acc[1][3]);
    }
#pragma unroll
    for (int i = 0; i < 2; i++)
#pragma unroll
        for (int j = 0; j < 4; j++)
#pragma unroll
            for (int v = 0; v < 8; v++) {
                int m = tm + i * 16 + v + half * 8;
                int n = tn + j * 16 + lm;
                C[m * Ncols + n] = f2bf(acc[i][j][v]);
            }
}

// ---------------- pass 2: per-row softmax stats (max, sum-exp) --------------
__global__ __launch_bounds__(256) void k_stats(const u16* __restrict__ Q,
                                               const u16* __restrict__ K,
                                               float* __restrict__ rowM,
                                               float* __restrict__ rowL) {
    const int wave = threadIdx.x >> 5;
    const int lane = threadIdx.x & 31;
    const int lm   = lane & 15;
    const int half = lane >> 4;
    const int lhA  = half * 8;
    const int lhB  = half * 16;

    const int idx = blockIdx.x * 8 + wave;          // 0..2047
    const int mt  = idx & 127;
    const int h   = (idx >> 7) & 7;
    const int b   = idx >> 10;

    const u16* qrow = Q + (b * NN + mt * 16 + lm) * QKC + h * NR;
    BfVec qa0, qa1;
    qa0.h[0] = *(const su8*)(qrow + lhA);
    qa0.h[1] = *(const su8*)(qrow + 16 + lhA);
    qa1.h[0] = *(const su8*)(qrow + 32 + lhA);
    qa1.h[1] = *(const su8*)(qrow + 48 + lhA);

    float runM[8], runS[8];
#pragma unroll
    for (int v = 0; v < 8; v++) { runM[v] = -1e30f; runS[v] = 0.f; }

    for (int nt = 0; nt < NN; nt += 16) {
        const u16* krow = K + (b * NN + nt + lm) * QKC + h * NR;
        BfVec k0, k1;
        k0.h[0] = *(const su8*)(krow + lhA);
        k0.h[1] = *(const su8*)(krow + 16 + lhA);
        k1.h[0] = *(const su8*)(krow + 32 + lhA);
        k1.h[1] = *(const su8*)(krow + 48 + lhA);
        v8f s = {};
        s = wmma_bf16(qa0, k0, s);
        s = wmma_bf16(qa1, k1, s);
#pragma unroll
        for (int v = 0; v < 8; v++) {
            float x = s[v];
            float m = fmaxf(runM[v], x);
            runS[v] = runS[v] * __expf(runM[v] - m) + __expf(x - m);
            runM[v] = m;
        }
    }
#pragma unroll
    for (int off = 1; off < 16; off <<= 1) {
#pragma unroll
        for (int v = 0; v < 8; v++) {
            float oM = __shfl_xor(runM[v], off);
            float oS = __shfl_xor(runS[v], off);
            float m  = fmaxf(runM[v], oM);
            runS[v]  = runS[v] * __expf(runM[v] - m) + oS * __expf(oM - m);
            runM[v]  = m;
        }
    }
    if (lm == 0) {
        float* Mp = rowM + (b * NH + h) * NN + mt * 16;
        float* Lp = rowL + (b * NH + h) * NN + mt * 16;
#pragma unroll
        for (int v = 0; v < 8; v++) {
            Mp[v + half * 8] = runM[v];
            Lp[v + half * 8] = runS[v];
        }
    }
}

// ---------------- pass 3: out[b,n,:] = sum_h P_h @ V_h ----------------------
// block = (b, 16-row tile); wave w owns d columns [w*128, w*128+128)
// K chunks staged to LDS via async-to-LDS, double buffered; V read from V^T
__global__ __launch_bounds__(256) void k_out(const u16* __restrict__ Q,
                                             const u16* __restrict__ Kg,
                                             const u16* __restrict__ VT,
                                             const float* __restrict__ rowM,
                                             const float* __restrict__ rowL,
                                             float* __restrict__ out) {
    __shared__ u16 kLDS[2][32][64];   // 8 KB, double-buffered K chunk
    __shared__ u16 pLDS[8][16][32];   // 8 KB, per-wave P tiles

    const int wave = threadIdx.x >> 5;
    const int lane = threadIdx.x & 31;
    const int lm   = lane & 15;
    const int half = lane >> 4;
    const int lhA  = half * 8;
    const int lhB  = half * 16;

    const int mt    = blockIdx.x & 127;
    const int b     = blockIdx.x >> 7;
    const int mbase = b * NN + mt * 16;
    const int dbase = wave * 128;

    // staging decomposition: one b128 per thread covers 32 rows x 64 cols
    const int srow = threadIdx.x >> 3;          // 0..31
    const int scol = (threadIdx.x & 7) * 8;     // 0..56

    v8f acc[8];
#pragma unroll
    for (int t = 0; t < 8; t++) acc[t] = {};

    int it = 0;
    for (int h = 0; h < NH; h++) {
        const u16* qrow = Q + (mbase + lm) * QKC + h * NR;
        BfVec qa0, qa1;
        qa0.h[0] = *(const su8*)(qrow + lhA);
        qa0.h[1] = *(const su8*)(qrow + 16 + lhA);
        qa1.h[0] = *(const su8*)(qrow + 32 + lhA);
        qa1.h[1] = *(const su8*)(qrow + 48 + lhA);

        float m8[8], il8[8];
        const float* Mp = rowM + (b * NH + h) * NN + mt * 16;
        const float* Lp = rowL + (b * NH + h) * NN + mt * 16;
#pragma unroll
        for (int v = 0; v < 8; v++) {
            int r  = v + half * 8;
            m8[v]  = Mp[r];
            il8[v] = 1.f / Lp[r];
        }

        for (int nt = 0; nt < NN; nt += 32) {
            const int buf = it & 1; it++;
            // ---- async-stage K[nt..nt+31][h*64..h*64+63] into LDS ----
            {
                const u16* src = Kg + (b * NN + nt + srow) * QKC + h * NR + scol;
                unsigned lds = (unsigned)(size_t)(&kLDS[buf][srow][scol]);
                unsigned long long ga = (unsigned long long)(size_t)src;
                asm volatile("global_load_async_to_lds_b128 %0, %1, off"
                             :: "v"(lds), "v"(ga) : "memory");
                asm volatile("s_wait_asynccnt 0" ::: "memory");
            }
            __syncthreads();

            // ---- scores for key columns nt..nt+31 (two 16-wide tiles) ----
            v8f s0 = {}, s1 = {};
            {
                BfVec k0, k1;
                k0.h[0] = *(const su8*)(&kLDS[buf][lm][lhA]);
                k0.h[1] = *(const su8*)(&kLDS[buf][lm][16 + lhA]);
                k1.h[0] = *(const su8*)(&kLDS[buf][lm][32 + lhA]);
                k1.h[1] = *(const su8*)(&kLDS[buf][lm][48 + lhA]);
                s0 = wmma_bf16(qa0, k0, s0);
                s0 = wmma_bf16(qa1, k1, s0);
            }
            {
                BfVec k0, k1;
                k0.h[0] = *(const su8*)(&kLDS[buf][16 + lm][lhA]);
                k0.h[1] = *(const su8*)(&kLDS[buf][16 + lm][16 + lhA]);
                k1.h[0] = *(const su8*)(&kLDS[buf][16 + lm][32 + lhA]);
                k1.h[1] = *(const su8*)(&kLDS[buf][16 + lm][48 + lhA]);
                s1 = wmma_bf16(qa0, k0, s1);
                s1 = wmma_bf16(qa1, k1, s1);
            }

            // ---- P = exp(S - M) / L, repack C-layout -> A-layout via LDS ----
#pragma unroll
            for (int v = 0; v < 8; v++) {
                int r = v + half * 8;
                pLDS[wave][r][lm]      = f2bf(__expf(s0[v] - m8[v]) * il8[v]);
                pLDS[wave][r][16 + lm] = f2bf(__expf(s1[v] - m8[v]) * il8[v]);
            }
            asm volatile("s_wait_dscnt 0" ::: "memory");  // wave-local LDS RAW
            BfVec pa;
            pa.h[0] = *(const su8*)(&pLDS[wave][lm][lhA]);
            pa.h[1] = *(const su8*)(&pLDS[wave][lm][16 + lhA]);

            // ---- O += P @ V : B operand straight from V^T, 2 b128/tile ----
            const u16* vt = VT + (h * ND + dbase + lm) * BN + b * NN + nt;
            __builtin_prefetch(vt + 32, 0, 0);
#pragma unroll
            for (int t = 0; t < 8; t++) {
                BfVec vB;
                vB.h[0] = *(const su8*)(vt + t * 16 * BN + lhA);
                vB.h[1] = *(const su8*)(vt + t * 16 * BN + 16 + lhA);
                acc[t] = wmma_bf16(pa, vB, acc[t]);
            }
        }
    }
#pragma unroll
    for (int t = 0; t < 8; t++)
#pragma unroll
        for (int v = 0; v < 8; v++)
            out[(mbase + v + half * 8) * ND + dbase + t * 16 + lm] = acc[t][v];
}

extern "C" void kernel_launch(void* const* d_in, const int* in_sizes, int n_in,
                              void* d_out, int out_size, void* d_ws, size_t ws_size,
                              hipStream_t stream) {
    (void)in_sizes; (void)n_in; (void)out_size; (void)ws_size;
    const float* x  = (const float*)d_in[0];   // [2,2048,1024]
    const float* Wq = (const float*)d_in[1];   // [512,1024]
    const float* Wk = (const float*)d_in[2];   // [512,1024]
    const float* Wv = (const float*)d_in[3];   // [8192,1024]

    char* ws = (char*)d_ws;
    size_t off = 0;
    u16* xb  = (u16*)(ws + off); off += (size_t)BN * 1024 * 2;       // 8 MB
    u16* wqb = (u16*)(ws + off); off += (size_t)QKC * 1024 * 2;      // 1 MB
    u16* wkb = (u16*)(ws + off); off += (size_t)QKC * 1024 * 2;      // 1 MB
    u16* wvb = (u16*)(ws + off); off += (size_t)VC * 1024 * 2;       // 16 MB
    u16* qb  = (u16*)(ws + off); off += (size_t)BN * QKC * 2;        // 4 MB
    u16* kb  = (u16*)(ws + off); off += (size_t)BN * QKC * 2;        // 4 MB
    u16* vbT = (u16*)(ws + off); off += (size_t)VC * BN * 2;         // 64 MB, V^T [8192][4096]
    float* rowM = (float*)(ws + off); off += (size_t)NB * NH * NN * 4;
    float* rowL = (float*)(ws + off); off += (size_t)NB * NH * NN * 4;

    // pass 0: conversions
    k_convert<<<1024, 256, 0, stream>>>(x,  xb,  BN * 1024);
    k_convert<<<256,  256, 0, stream>>>(Wq, wqb, QKC * 1024);
    k_convert<<<256,  256, 0, stream>>>(Wk, wkb, QKC * 1024);
    k_convert<<<2048, 256, 0, stream>>>(Wv, wvb, VC * 1024);

    // pass 1: projections. macro-tiles are 32x64, 8 per block.
    // Q,K: [4096 x 512];  V^T = Wv @ x^T: [8192 x 4096] (operand roles swapped)
    k_gemm<<<(BN / 32) * (QKC / 64) / 8, 256, 0, stream>>>(xb,  wqb, qb,  QKC, 1024);
    k_gemm<<<(BN / 32) * (QKC / 64) / 8, 256, 0, stream>>>(xb,  wkb, kb,  QKC, 1024);
    k_gemm<<<(VC / 32) * (BN  / 64) / 8, 256, 0, stream>>>(wvb, xb,  vbT, BN,  1024);

    // pass 2: softmax row stats
    k_stats<<<256, 256, 0, stream>>>(qb, kb, rowM, rowL);

    // pass 3: attention output, head-summed
    k_out<<<NB * (NN / 16), 256, 0, stream>>>(qb, kb, vbT, rowM, rowL, (float*)d_out);
}
